// GeometryBlock_44976897524202
// MI455X (gfx1250) — matrix operand order repaired and checked by reading, not compiled, not run on hardware
//
#include <hip/hip_runtime.h>
#include <hip/hip_bf16.h>
#include <math.h>

#define BB 4
#define LL 2048
#define DD 256
#define NSETS 4
#define PP 4
#define POSP 16
#define LTMP 16

static __device__ __forceinline__ float sigf(float x) { return 1.0f / (1.0f + expf(-x)); }

constexpr float PI_F = 3.14159265358979323846f;

typedef __attribute__((ext_vector_type(16))) _Float16 v16h;
typedef __attribute__((ext_vector_type(8)))  _Float16 v8h;
typedef __attribute__((ext_vector_type(8)))  float    v8f;

union Frag { v16h v; v8h h8[2]; };

// ---------------------------------------------------------------------------
// fp32 -> fp16 elementwise convert (coalesced, 4 elems/thread)
// ---------------------------------------------------------------------------
__global__ void cvt_f16_kernel(const float* __restrict__ src,
                               _Float16* __restrict__ dst, int n4) {
  int i = (blockIdx.x * blockDim.x + threadIdx.x);
  if (i >= n4) return;
  const float4 v = ((const float4*)src)[i];
  v8h* d = (v8h*)dst;                 // write 4 halves via 64-bit store
  _Float16 tmp[4] = {(_Float16)v.x, (_Float16)v.y, (_Float16)v.z, (_Float16)v.w};
  ((ushort1*)dst)[0];                 // no-op keep types simple
  _Float16* o = dst + (size_t)i * 4;
  o[0] = tmp[0]; o[1] = tmp[1]; o[2] = tmp[2]; o[3] = tmp[3];
  (void)d;
}

// ---------------------------------------------------------------------------
// W[K,N] fp32 -> Wt[N,K] fp16 (transpose + convert; coalesced reads)
// ---------------------------------------------------------------------------
__global__ void cvt_transpose_kernel(const float* __restrict__ W,
                                     _Float16* __restrict__ Wt, int K, int N) {
  int idx = blockIdx.x * blockDim.x + threadIdx.x;
  if (idx >= K * N) return;
  int k = idx / N, n = idx % N;
  Wt[(size_t)n * K + k] = (_Float16)W[idx];
}

// ---------------------------------------------------------------------------
// Kernel 1: per-batch running mean / std scan (emits fp16 for LTM GEMM)
// ---------------------------------------------------------------------------
__global__ void stats_kernel(const float* __restrict__ x,
                             _Float16* __restrict__ rm16,
                             _Float16* __restrict__ rs16) {
  const int b = blockIdx.x;
  const int d = threadIdx.x;
  const float* xb = x + (size_t)b * LL * DD;
  float s = 0.0f, s2 = 0.0f;
  for (int t = 0; t < LL; ++t) {
    float v = xb[(size_t)t * DD + d];
    s += v; s2 += v * v;
    float inv = 1.0f / (float)(t + 1);
    float m = s * inv;
    float var = s2 * inv - m * m;
    rm16[((size_t)b * LL + t) * DD + d] = (_Float16)m;
    rs16[((size_t)b * LL + t) * DD + d] = (_Float16)sqrtf(fmaxf(var, 1e-8f));
  }
}

// ---------------------------------------------------------------------------
// WMMA GEMM on pre-converted fp16 operands.
//   A_cat = concat of NSEG row-major [M,256] fp16 matrices
//   Wt    = [N,K] fp16 (pre-transposed)
//   mode 0: out = acc + bias + res   (fp32 epilogue; bias/res nullable)
//   mode 1: out = tanh(acc) * pi
// One 16x16 tile per wave32. Per K-step: 2x b128 A loads + 2x b128 B loads
// feeding v_wmma_f32_16x16x32_f16 directly (no conversions in the loop).
// Fragment layouts per CDNA5 ISA 7.12.2:
//   A 16x32 f16 : elem e -> K = (e&7) + (e>>3)*16 + (lane/16)*8   (two 8-half runs)
//   B 32x16 f16 : elem e -> K = (e&7) + (e>>3)*16 + (lane/16)*8   (two 8-half runs)
//   C/D f32     : vgpr r -> M = r + (lane/16)*8, N = lane&15
// ---------------------------------------------------------------------------
template <int NSEG>
__global__ void wmma_gemm_f16_kernel(const _Float16* __restrict__ A0,
                                     const _Float16* __restrict__ A1,
                                     const _Float16* __restrict__ A2,
                                     const _Float16* __restrict__ Wt,
                                     const float* __restrict__ bias,
                                     const float* __restrict__ res,
                                     float* __restrict__ out,
                                     int M, int N, int mode) {
  const int K = NSEG * 256;
  const int lane = threadIdx.x & 31;
  const int wave = threadIdx.x >> 5;
  const int gw   = blockIdx.x * (blockDim.x >> 5) + wave;
  const int NT   = N >> 4;
  const int tile_m = gw / NT;
  const int tile_n = gw % NT;
  if (tile_m * 16 >= M) return;          // wave-uniform: EXEC all-ones at WMMA
  const int half = lane >> 4;
  const int mrow = tile_m * 16 + (lane & 15);
  const int ncol = tile_n * 16 + (lane & 15);

  const _Float16* Wrow = Wt + (size_t)ncol * K;

  v8f c = {0.f, 0.f, 0.f, 0.f, 0.f, 0.f, 0.f, 0.f};

#pragma unroll
  for (int kk = 0; kk < K; kk += 32) {
    Frag a, b;
    // A: two contiguous 8-half (16B) runs at K = kk+half*8 and +16
    {
      int ka0 = kk + half * 8;
      int ka1 = ka0 + 16;
      const _Float16* Ap0 = (NSEG == 1) ? A0
                          : ((ka0 >> 8) == 0 ? A0 : ((ka0 >> 8) == 1 ? A1 : A2));
      const _Float16* Ap1 = (NSEG == 1) ? A0
                          : ((ka1 >> 8) == 0 ? A0 : ((ka1 >> 8) == 1 ? A1 : A2));
      a.h8[0] = *(const v8h*)(Ap0 + (size_t)mrow * 256 + (ka0 & 255));
      a.h8[1] = *(const v8h*)(Ap1 + (size_t)mrow * 256 + (ka1 & 255));
    }
    // B: two contiguous 8-half (16B) runs at K = kk+half*8 and +16
    {
      const _Float16* Wp = Wrow + kk + half * 8;
      b.h8[0] = *(const v8h*)(Wp);
      b.h8[1] = *(const v8h*)(Wp + 16);
    }
    c = __builtin_amdgcn_wmma_f32_16x16x32_f16(false, a.v, false, b.v,
                                               (short)0, c, false, false);
  }

#pragma unroll
  for (int r = 0; r < 8; ++r) {
    int m = tile_m * 16 + r + half * 8;
    float val = c[r];
    if (bias) val += bias[ncol];
    if (mode == 1) val = tanhf(val) * PI_F;
    if (res) val += res[(size_t)m * N + ncol];
    out[(size_t)m * N + ncol] = val;
  }
}

// ---------------------------------------------------------------------------
// Kernel 3: sequential holographic-memory scan. One workgroup per batch,
// 256 threads (8 wave32s); thread d owns channel d. All complex memory state
// in VGPRs; per-token shared scalars via LDS. hn emitted in fp16 for the
// final WMMA GEMM. Next token's V row is prefetched (global_prefetch_b8).
// ---------------------------------------------------------------------------
__global__ void scan_kernel(const float* __restrict__ V,
                            const float* __restrict__ kang,
                            const float* __restrict__ qang,
                            const float* __restrict__ lang,
                            const float* __restrict__ pos_freqs,
                            const float* __restrict__ ltm_mem,   // complex64 pairs [LTMP][D]
                            const float* __restrict__ set_weights,
                            const float* __restrict__ pos_weight,
                            const float* __restrict__ surprise_scale,
                            const float* __restrict__ surprise_bias,
                            const float* __restrict__ resonance_scale,
                            const float* __restrict__ resonance_threshold,
                            const float* __restrict__ ltm_weight,
                            const float* __restrict__ ltm_count,
                            const float* __restrict__ ln_gamma,
                            const float* __restrict__ ln_beta,
                            _Float16* __restrict__ hn16_out) {
  const int b   = blockIdx.x;
  const int tid = threadIdx.x;
  const int d   = tid;
  const int lane = tid & 31;
  const int wid  = tid >> 5;

  __shared__ float kc[16], ksn[16], qc[16], qsn[16], lc[16], lsn[16], pc[16], psn[16];
  __shared__ float jkr[4], jki[4], jqr[4], jqi[4];
  __shared__ float kmr[4], kmi[4];
  __shared__ float gateS;
  __shared__ float wsum[8], wsum2[8];
  __shared__ float muS, rinvS;

  if (tid < 4) { kmr[tid] = 0.0f; kmi[tid] = 0.0f; }

  // per-thread broadcast constants
  float e0 = expf(set_weights[0]), e1 = expf(set_weights[1]);
  float e2 = expf(set_weights[2]), e3 = expf(set_weights[3]);
  float esum = e0 + e1 + e2 + e3;
  float sw[4] = {e0 / esum, e1 / esum, e2 / esum, e3 / esum};
  const float sigPos = sigf(pos_weight[0]);
  const float sigLtm = sigf(ltm_weight[0]);
  const float pnorm  = sqrtf(fmaxf(ltm_count[0], 1.0f) * (float)LTMP);
  const float sscale = surprise_scale[0];
  const float sbias  = surprise_bias[0];
  const float rsc    = fminf(fmaxf(resonance_scale[0], 1.0f), 20.0f);
  const float rth    = fminf(fmaxf(resonance_threshold[0], 0.1f), 0.9f);
  const float gam    = ln_gamma[d];
  const float bet    = ln_beta[d];

  // register-resident complex memories
  float cmr[4], cmi[4], bmr[16], bmi[16], pmr[16], pmi[16], lmr[16], lmi[16];
#pragma unroll
  for (int p = 0; p < 4; ++p) { cmr[p] = 0.0f; cmi[p] = 0.0f; }
#pragma unroll
  for (int p = 0; p < 16; ++p) {
    bmr[p] = 0.0f; bmi[p] = 0.0f; pmr[p] = 0.0f; pmi[p] = 0.0f;
    lmr[p] = ltm_mem[((size_t)p * DD + d) * 2 + 0];
    lmi[p] = ltm_mem[((size_t)p * DD + d) * 2 + 1];
  }
  __syncthreads();

  const size_t rowbase = (size_t)b * LL;

  for (int t = 0; t < LL; ++t) {
    // prefetch next token's V row toward the WGP (global_prefetch_b8)
    if (t + 1 < LL) __builtin_prefetch(V + (rowbase + t + 1) * DD + d, 0, 0);

    // phase 1: plane phasors
    if (tid < 64) {
      int grp = tid >> 4, p = tid & 15;
      float ang;
      if (grp == 0)      ang = kang[(rowbase + t) * 16 + p];
      else if (grp == 1) ang = qang[(rowbase + t) * 16 + p];
      else if (grp == 2) ang = lang[(rowbase + t) * 16 + p];
      else               ang = (float)t * pos_freqs[p] * (2.0f * PI_F);
      float sv, cv;
      sincosf(ang, &sv, &cv);
      if (grp == 0)      { kc[p] = cv; ksn[p] = sv; }
      else if (grp == 1) { qc[p] = cv; qsn[p] = sv; }
      else if (grp == 2) { lc[p] = cv; lsn[p] = sv; }
      else               { pc[p] = cv; psn[p] = sv; }
    }
    __syncthreads();

    // phase 2: joint (cross-bank) phasors jk, jq
    if (tid < 8) {
      int p = tid & 3;
      const float* cs = (tid < 4) ? kc : qc;
      const float* sn = (tid < 4) ? ksn : qsn;
      float pr = 1.0f, pi = 0.0f;
#pragma unroll
      for (int s = 0; s < 4; ++s) {
        float cr = cs[s * 4 + p], ci = sn[s * 4 + p];
        float nr = pr * cr - pi * ci;
        pi = pr * ci + pi * cr;
        pr = nr;
      }
      if (tid < 4) { jkr[p] = pr; jki[p] = pi; }
      else         { jqr[p] = pr; jqi[p] = pi; }
    }
    __syncthreads();

    // phase 3: resonance gate + exclusive km update (thread 0)
    if (tid == 0) {
      float rmag = 0.0f;
#pragma unroll
      for (int p = 0; p < 4; ++p) {
        float ar = kmr[p] * jqr[p] + kmi[p] * jqi[p];
        float ai = kmi[p] * jqr[p] - kmr[p] * jqi[p];
        rmag += sqrtf(ar * ar + ai * ai);
      }
      rmag *= 0.25f;
      float nres = rmag * rsqrtf(fmaxf((float)t, 1.0f));
      float sur  = 0.5f * (1.0f - tanhf(rsc * (nres - rth)));
      gateS = sigf(sscale * (sur - 0.5f) + sbias);
#pragma unroll
      for (int p = 0; p < 4; ++p) { kmr[p] += jkr[p]; kmi[p] += jki[p]; }
    }
    __syncthreads();

    // phase 4: per-channel memory update + retrieval
    const float gate = gateS;
    const float v  = V[(rowbase + t) * DD + d];
    const float vg = v * gate;

    float crossRet = 0.0f;
#pragma unroll
    for (int p = 0; p < 4; ++p) {
      cmr[p] += jkr[p] * vg;
      cmi[p] += jki[p] * vg;
      crossRet += cmr[p] * jqr[p] + cmi[p] * jqi[p];   // Re(mem * conj(jq))
    }

    float tb = 0.0f;
#pragma unroll
    for (int s = 0; s < 4; ++s) {
      float per = 0.0f;
#pragma unroll
      for (int p = 0; p < 4; ++p) {
        int i = s * 4 + p;
        bmr[i] += kc[i] * vg;
        bmi[i] += ksn[i] * vg;
        per += bmr[i] * qc[i] + bmi[i] * qsn[i];
      }
      tb += sw[s] * per;
    }

    float rp = 0.0f;
#pragma unroll
    for (int p = 0; p < 16; ++p) {
      pmr[p] += pc[p] * v;           // positional memory uses ungated V
      pmi[p] += psn[p] * v;
      rp += pmr[p] * pc[p] + pmi[p] * psn[p];
    }

    float pers = 0.0f;
#pragma unroll
    for (int p = 0; p < 16; ++p) pers += lmr[p] * lc[p] + lmi[p] * lsn[p];

    float total = (tb + crossRet) / (float)(NSETS + 1);
    total = 0.5f * total + 0.5f * (sigPos * rp);
    total += sigLtm * (pers / pnorm);
    float h = total * rsqrtf((float)(t + 1) * (float)PP);

    // phase 5: LayerNorm stats (wave32 shuffles + LDS combine)
    float s1 = h, s2 = h * h;
#pragma unroll
    for (int off = 16; off >= 1; off >>= 1) {
      s1 += __shfl_xor(s1, off, 32);
      s2 += __shfl_xor(s2, off, 32);
    }
    if (lane == 0) { wsum[wid] = s1; wsum2[wid] = s2; }
    __syncthreads();
    if (tid == 0) {
      float a = 0.0f, q = 0.0f;
#pragma unroll
      for (int w = 0; w < 8; ++w) { a += wsum[w]; q += wsum2[w]; }
      float mu  = a / (float)DD;
      float var = q / (float)DD - mu * mu;
      muS = mu;
      rinvS = rsqrtf(var + 1e-5f);
    }
    __syncthreads();

    hn16_out[(rowbase + t) * DD + d] = (_Float16)((h - muS) * rinvS * gam + bet);
  }
}

// ---------------------------------------------------------------------------
// Launcher
// ---------------------------------------------------------------------------
extern "C" void kernel_launch(void* const* d_in, const int* in_sizes, int n_in,
                              void* d_out, int out_size, void* d_ws, size_t ws_size,
                              hipStream_t stream) {
  (void)in_sizes; (void)n_in; (void)out_size; (void)ws_size;

  const float* x            = (const float*)d_in[0];
  const float* key_proj     = (const float*)d_in[1];
  const float* query_proj   = (const float*)d_in[2];
  const float* ltm_key_proj = (const float*)d_in[3];
  const float* pos_freqs    = (const float*)d_in[4];
  const float* w_value      = (const float*)d_in[5];
  const float* b_value      = (const float*)d_in[6];
  // d_in[7] w_ltm_val, d_in[8] b_ltm_val : unused by reference
  const float* ln_gamma     = (const float*)d_in[9];
  const float* ln_beta      = (const float*)d_in[10];
  const float* w_out        = (const float*)d_in[11];
  const float* b_out        = (const float*)d_in[12];
  const float* set_weights  = (const float*)d_in[13];
  const float* pos_weight   = (const float*)d_in[14];
  const float* surprise_scale      = (const float*)d_in[15];
  const float* surprise_bias       = (const float*)d_in[16];
  const float* resonance_scale     = (const float*)d_in[17];
  const float* resonance_threshold = (const float*)d_in[18];
  const float* ltm_weight   = (const float*)d_in[19];
  const float* ltm_mem      = (const float*)d_in[20];   // complex64 -> float pairs
  const float* ltm_count    = (const float*)d_in[21];

  float* out = (float*)d_out;

  const size_t M = (size_t)BB * LL;        // 8192 rows

  // workspace carve-up (256B aligned)
  char* wp = (char*)d_ws;
  auto carve = [&wp](size_t bytes) {
    char* p = wp;
    wp += (bytes + 255) & ~(size_t)255;
    return p;
  };
  float*    V    = (float*)   carve(M * DD * sizeof(float));
  float*    kang = (float*)   carve(M * 16 * sizeof(float));
  float*    qang = (float*)   carve(M * 16 * sizeof(float));
  float*    lang = (float*)   carve(M * 16 * sizeof(float));
  _Float16* x16  = (_Float16*)carve(M * DD * sizeof(_Float16));
  _Float16* rm16 = (_Float16*)carve(M * DD * sizeof(_Float16));
  _Float16* rs16 = (_Float16*)carve(M * DD * sizeof(_Float16));
  _Float16* hn16 = (_Float16*)carve(M * DD * sizeof(_Float16));
  _Float16* kp_t = (_Float16*)carve((size_t)16 * 256 * sizeof(_Float16));
  _Float16* qp_t = (_Float16*)carve((size_t)16 * 256 * sizeof(_Float16));
  _Float16* lp_t = (_Float16*)carve((size_t)16 * 768 * sizeof(_Float16));
  _Float16* wv_t = (_Float16*)carve((size_t)256 * 256 * sizeof(_Float16));
  _Float16* wo_t = (_Float16*)carve((size_t)256 * 256 * sizeof(_Float16));

  // 0) operand preparation: x -> fp16; weights -> transposed fp16
  {
    int n4 = (int)(M * DD / 4);
    cvt_f16_kernel<<<(n4 + 255) / 256, 256, 0, stream>>>(x, x16, n4);
    cvt_transpose_kernel<<<(256 * 16 + 255) / 256, 256, 0, stream>>>(key_proj, kp_t, 256, 16);
    cvt_transpose_kernel<<<(256 * 16 + 255) / 256, 256, 0, stream>>>(query_proj, qp_t, 256, 16);
    cvt_transpose_kernel<<<(768 * 16 + 255) / 256, 256, 0, stream>>>(ltm_key_proj, lp_t, 768, 16);
    cvt_transpose_kernel<<<(256 * 256 + 255) / 256, 256, 0, stream>>>(w_value, wv_t, 256, 256);
    cvt_transpose_kernel<<<(256 * 256 + 255) / 256, 256, 0, stream>>>(w_out, wo_t, 256, 256);
  }

  // 1) running stats scan (emits fp16 rm/rs)
  stats_kernel<<<BB, DD, 0, stream>>>(x, rm16, rs16);

  // 2) WMMA GEMMs: projections + value
  {
    int blocks = ((int)M / 16) / 8;                 // N = 16 -> 1 tile per row-tile
    wmma_gemm_f16_kernel<1><<<blocks, 256, 0, stream>>>(x16, nullptr, nullptr,
        kp_t, nullptr, nullptr, kang, (int)M, 16, 1);
    wmma_gemm_f16_kernel<1><<<blocks, 256, 0, stream>>>(x16, nullptr, nullptr,
        qp_t, nullptr, nullptr, qang, (int)M, 16, 1);
    wmma_gemm_f16_kernel<3><<<blocks, 256, 0, stream>>>(x16, rm16, rs16,
        lp_t, nullptr, nullptr, lang, (int)M, 16, 1);
  }
  {
    int blocks = (((int)M / 16) * (DD / 16)) / 8;   // N = 256
    wmma_gemm_f16_kernel<1><<<blocks, 256, 0, stream>>>(x16, nullptr, nullptr,
        wv_t, b_value, nullptr, V, (int)M, DD, 0);
  }

  // 3) sequential holographic scan -> hn (fp16)
  scan_kernel<<<BB, DD, 0, stream>>>(V, kang, qang, lang, pos_freqs, ltm_mem,
      set_weights, pos_weight, surprise_scale, surprise_bias,
      resonance_scale, resonance_threshold, ltm_weight, ltm_count,
      ln_gamma, ln_beta, hn16);

  // 4) out = x + hn @ w_out + b_out   (WMMA, residual fused)
  {
    int blocks = (((int)M / 16) * (DD / 16)) / 8;
    wmma_gemm_f16_kernel<1><<<blocks, 256, 0, stream>>>(hn16, nullptr, nullptr,
        wo_t, b_out, x, out, (int)M, DD, 0);
  }
}
